// DetrMultiHeadDotProductAttention_20856361189640
// MI455X (gfx1250) — matrix operand order repaired and checked by reading, not compiled, not run on hardware
//
#include <hip/hip_runtime.h>
#include <stdint.h>

typedef __attribute__((ext_vector_type(16))) _Float16 v16h;
typedef __attribute__((ext_vector_type(8)))  float    v8f;
typedef int v4i_ __attribute__((vector_size(16)));   // matches builtin param type

#define B_  4
#define S_  1024
#define D_  1024
#define H_  16
#define HD_ 64

// Async global->LDS copies (CDNA5): guarded so a missing builtin falls back
// to the synchronous VGPR staging path instead of breaking the build.
#if defined(__HIP_DEVICE_COMPILE__) && defined(__has_builtin)
#if __has_builtin(__builtin_amdgcn_global_load_async_to_lds_b128)
#define USE_ASYNC_LDS 1
#endif
#endif
#ifndef USE_ASYNC_LDS
#define USE_ASYNC_LDS 0
#endif

// generic->AS(1): full 64-bit value; generic->AS(3): low 32 bits are the LDS
// byte offset per the flat-address aperture mapping (ISA 10.2).
#define AS1P(p) ((__attribute__((address_space(1))) v4i_*)(uintptr_t)(p))
#define AS3P(p) ((__attribute__((address_space(3))) v4i_*)(uint32_t)(uintptr_t)(p))

union FragH { v16h v; uint4 q[2]; };
union Half8 { uint4 q; _Float16 h[8]; };
struct __align__(8) H4 { _Float16 h[4]; };

static __device__ __forceinline__ v8f wmma_f16(v16h a, v16h b, v8f c) {
  return __builtin_amdgcn_wmma_f32_16x16x32_f16(false, a, false, b, (short)0, c,
                                                false, false);
}

static __device__ __forceinline__ void wait_async0() {
  asm volatile("s_wait_asynccnt 0x0" ::: "memory");
}

// ---------------------------------------------------------------------------
// Kernel 1: q_in = inputs_q + pos_q ; k_in = inputs_kv + pos_k ; v_in = inputs_kv + pos_v
// f32 -> f16, vectorized.
// ---------------------------------------------------------------------------
__global__ __launch_bounds__(256) void prep_qkv(
    const float* __restrict__ inq, const float* __restrict__ inkv,
    const float* __restrict__ pq,  const float* __restrict__ pk,
    const float* __restrict__ pv,
    _Float16* __restrict__ qh, _Float16* __restrict__ kh, _Float16* __restrict__ vh) {
  int i = blockIdx.x * 256 + threadIdx.x;          // float4 index, 1M total
  float4 xq = ((const float4*)inq)[i];
  float4 xk = ((const float4*)inkv)[i];
  float4 eq = ((const float4*)pq)[i];
  float4 ek = ((const float4*)pk)[i];
  float4 ev = ((const float4*)pv)[i];
  H4 oq, ok, ov;
  oq.h[0] = (_Float16)(xq.x + eq.x); oq.h[1] = (_Float16)(xq.y + eq.y);
  oq.h[2] = (_Float16)(xq.z + eq.z); oq.h[3] = (_Float16)(xq.w + eq.w);
  ok.h[0] = (_Float16)(xk.x + ek.x); ok.h[1] = (_Float16)(xk.y + ek.y);
  ok.h[2] = (_Float16)(xk.z + ek.z); ok.h[3] = (_Float16)(xk.w + ek.w);
  ov.h[0] = (_Float16)(xk.x + ev.x); ov.h[1] = (_Float16)(xk.y + ev.y);
  ov.h[2] = (_Float16)(xk.z + ev.z); ov.h[3] = (_Float16)(xk.w + ev.w);
  ((H4*)qh)[i] = oq;
  ((H4*)kh)[i] = ok;
  ((H4*)vh)[i] = ov;
}

// ---------------------------------------------------------------------------
// Kernel 2: weight transpose + f32->f16: Wt[n][k] = W[k][n], 1024x1024.
// ---------------------------------------------------------------------------
__global__ __launch_bounds__(256) void transpose_w(const float* __restrict__ in,
                                                   _Float16* __restrict__ out) {
  __shared__ float tile[32][33];
  int tx = threadIdx.x, ty = threadIdx.y;
  int x  = blockIdx.x * 32 + tx;
  int yb = blockIdx.y * 32;
#pragma unroll
  for (int r = 0; r < 4; r++)
    tile[ty + r * 8][tx] = in[(size_t)(yb + ty + r * 8) * D_ + x];
  __syncthreads();
#pragma unroll
  for (int r = 0; r < 4; r++)
    out[(size_t)(blockIdx.x * 32 + ty + r * 8) * D_ + yb + tx] =
        (_Float16)tile[tx][ty + r * 8];
}

// ---------------------------------------------------------------------------
// Kernel 3: C[4096,1024] = A[4096,1024] * Bt^T  (Bt stored [N][K] f16)
// 128x128x32 tiles, 8 waves, wave = 32x64 (2x4 WMMA tiles).
// Async double-buffered LDS staging when available.
// MODE 0: out f16 permuted to [B,H,S,HD], val=(acc+bias)*scale
// MODE 1: out f32 [M,N], val=acc+bias
// ---------------------------------------------------------------------------
template <int MODE>
__global__ __launch_bounds__(256) void gemm_f16(
    const _Float16* __restrict__ A, const _Float16* __restrict__ Bt,
    const float* __restrict__ bias, float scale,
    _Float16* __restrict__ outH, float* __restrict__ outF) {
#if USE_ASYNC_LDS
  __shared__ _Float16 Alds[2][128 * 48];   // 96B row stride (6x16B)
  __shared__ _Float16 Blds[2][128 * 48];
#else
  __shared__ _Float16 Alds[1][128 * 48];
  __shared__ _Float16 Blds[1][128 * 48];
#endif
  const int t = threadIdx.x, lane = t & 31, w = t >> 5;
  const int wm = w & 3, wn = w >> 2;
  const int bm = blockIdx.y * 128, bn = blockIdx.x * 128;

  v8f acc[2][4];
  v8f vz = {0.f, 0.f, 0.f, 0.f, 0.f, 0.f, 0.f, 0.f};
#pragma unroll
  for (int i = 0; i < 2; i++)
#pragma unroll
    for (int j = 0; j < 4; j++) acc[i][j] = vz;

  auto compute = [&](int buf) {
    FragH af[2];
#pragma unroll
    for (int i = 0; i < 2; i++) {   // A frag: lane<16 holds K{0..7,16..23}
      int row = wm * 32 + i * 16 + (lane & 15);
      int o1  = (lane >> 4) << 3;
      af[i].q[0] = *(const uint4*)&Alds[buf][row * 48 + o1];
      af[i].q[1] = *(const uint4*)&Alds[buf][row * 48 + o1 + 16];
    }
    FragH bf[4];
#pragma unroll
    for (int j = 0; j < 4; j++) {   // B frag: lane<16 holds col n, K{0..7,16..23}
      int n  = wn * 64 + j * 16 + (lane & 15);
      int ko = (lane >> 4) << 3;
      bf[j].q[0] = *(const uint4*)&Blds[buf][n * 48 + ko];
      bf[j].q[1] = *(const uint4*)&Blds[buf][n * 48 + ko + 16];
    }
#pragma unroll
    for (int i = 0; i < 2; i++)
#pragma unroll
      for (int j = 0; j < 4; j++)
        acc[i][j] = wmma_f16(af[i].v, bf[j].v, acc[i][j]);
  };

#if USE_ASYNC_LDS
  auto stage = [&](int buf, int kb) {
#pragma unroll
    for (int c = t; c < 512; c += 256) {
      int row = c >> 2, off = (c & 3) << 3;   // 16B chunks
      __builtin_amdgcn_global_load_async_to_lds_b128(
          AS1P(&A[(size_t)(bm + row) * D_ + kb + off]),
          AS3P(&Alds[buf][row * 48 + off]), 0, 0);
      __builtin_amdgcn_global_load_async_to_lds_b128(
          AS1P(&Bt[(size_t)(bn + row) * D_ + kb + off]),
          AS3P(&Blds[buf][row * 48 + off]), 0, 0);
    }
  };
  stage(0, 0);
#pragma unroll 1
  for (int i = 0; i < 32; i++) {
    wait_async0();          // copies for buf[i&1] have landed
    __syncthreads();        // everyone done with buf[(i+1)&1] from iter i-1
    if (i + 1 < 32) stage((i + 1) & 1, (i + 1) * 32);
    compute(i & 1);
  }
#else
#pragma unroll 1
  for (int kb = 0; kb < D_; kb += 32) {
    __syncthreads();
#pragma unroll
    for (int c = t; c < 512; c += 256) {
      int row = c >> 2, off = (c & 3) << 3;
      *(uint4*)&Alds[0][row * 48 + off] =
          *(const uint4*)&A[(size_t)(bm + row) * D_ + kb + off];
      *(uint4*)&Blds[0][row * 48 + off] =
          *(const uint4*)&Bt[(size_t)(bn + row) * D_ + kb + off];
    }
    if (kb + 32 < D_)
      __builtin_prefetch(&A[(size_t)(bm + (t >> 2)) * D_ + kb + 32], 0, 1);
    __syncthreads();
    compute(0);
  }
#endif

#pragma unroll
  for (int i = 0; i < 2; i++)
#pragma unroll
    for (int j = 0; j < 4; j++) {
      int n = bn + wn * 64 + j * 16 + (lane & 15);
      float bv = bias[n];
#pragma unroll
      for (int g = 0; g < 8; g++) {
        int m = bm + wm * 32 + i * 16 + g + ((lane >> 4) << 3);
        float val = (acc[i][j][g] + bv) * scale;
        if (MODE == 0) {
          int bb = m >> 10, ss = m & 1023, hh = n >> 6, dd = n & 63;
          outH[((size_t)((bb * H_ + hh) << 10) + ss) * HD_ + dd] = (_Float16)val;
        } else {
          outF[(size_t)m * D_ + n] = val;
        }
      }
    }
}

// ---------------------------------------------------------------------------
// Kernel 4: flash attention per (b, h, 128 q-rows). Key blocks of 64.
// X written as [b, s, h*64+d] f16 for the output projection.
// ---------------------------------------------------------------------------
__global__ __launch_bounds__(256) void attn_fused(
    const _Float16* __restrict__ Qp, const _Float16* __restrict__ Kp,
    const _Float16* __restrict__ Vp, const float* __restrict__ mask,
    _Float16* __restrict__ X) {
  __shared__ _Float16 Klds[64 * 80];       // [key][d], 160B stride
  __shared__ _Float16 Vtlds[64 * 72];      // [d][key], 144B stride
  __shared__ _Float16 Plds[8 * 16 * 72];   // per-wave P scratch, 144B stride
  const int t = threadIdx.x, lane = t & 31, w = t >> 5;
  const int qt = blockIdx.x, h = blockIdx.y, b = blockIdx.z;
  const size_t base = (size_t)(b * H_ + h) * S_ * HD_;
  const _Float16* Qb = Qp + base;
  const _Float16* Kb = Kp + base;
  const _Float16* Vb = Vp + base;

  // Q fragments loaded once, directly in A-fragment layout.
  const int qr = qt * 128 + w * 16 + (lane & 15);
  FragH aq[2];
#pragma unroll
  for (int tt = 0; tt < 2; tt++) {
    int o1 = tt * 32 + ((lane >> 4) << 3);
    aq[tt].q[0] = *(const uint4*)&Qb[(size_t)qr * HD_ + o1];
    aq[tt].q[1] = *(const uint4*)&Qb[(size_t)qr * HD_ + o1 + 16];
  }

  float m_run[8], l_run[8];
  v8f oacc[4];
  v8f vz = {0.f, 0.f, 0.f, 0.f, 0.f, 0.f, 0.f, 0.f};
#pragma unroll
  for (int g = 0; g < 8; g++) { m_run[g] = -3.0e38f; l_run[g] = 0.f; }
#pragma unroll
  for (int jn = 0; jn < 4; jn++) oacc[jn] = vz;

  _Float16* Pw = &Plds[w * 16 * 72];

#pragma unroll 1
  for (int kb = 0; kb < 16; kb++) {
    __syncthreads();
    const _Float16* Kblk = Kb + (size_t)kb * 64 * HD_;
    const _Float16* Vblk = Vb + (size_t)kb * 64 * HD_;
#if USE_ASYNC_LDS
#pragma unroll
    for (int c = t; c < 512; c += 256) {          // K: async row-major copy
      int row = c >> 3, off = (c & 7) << 3;
      __builtin_amdgcn_global_load_async_to_lds_b128(
          AS1P(&Kblk[row * HD_ + off]), AS3P(&Klds[row * 80 + off]), 0, 0);
    }
#else
#pragma unroll
    for (int c = t; c < 512; c += 256) {          // K: row-major copy
      int row = c >> 3, off = (c & 7) << 3;
      *(uint4*)&Klds[row * 80 + off] = *(const uint4*)&Kblk[row * HD_ + off];
    }
#endif
#pragma unroll
    for (int c = t; c < 512; c += 256) {          // V: transpose into LDS
      int key = c >> 3, d0 = (c & 7) << 3;
      Half8 vv; vv.q = *(const uint4*)&Vblk[key * HD_ + d0];
#pragma unroll
      for (int jj = 0; jj < 8; jj++) Vtlds[(d0 + jj) * 72 + key] = vv.h[jj];
    }
#if USE_ASYNC_LDS
    wait_async0();
#endif
    __syncthreads();

    float kbias[4];
#pragma unroll
    for (int j = 0; j < 4; j++)
      kbias[j] = (1.0f - mask[b * S_ + kb * 64 + j * 16 + (lane & 15)]) * -1.0e10f;

    // scores S = Q K^T  (16 x 64 per wave)
    v8f s[4];
#pragma unroll
    for (int j = 0; j < 4; j++) {
      v8f a = vz;
#pragma unroll
      for (int tt = 0; tt < 2; tt++) {
        FragH bk;
        int n  = j * 16 + (lane & 15);
        int ko = tt * 32 + ((lane >> 4) << 3);
        bk.q[0] = *(const uint4*)&Klds[n * 80 + ko];
        bk.q[1] = *(const uint4*)&Klds[n * 80 + ko + 16];
        a = wmma_f16(aq[tt].v, bk.v, a);
      }
#pragma unroll
      for (int g = 0; g < 8; g++) a[g] += kbias[j];
      s[j] = a;
    }

    // online softmax (rows g live per 16-lane half)
    float tm[8];
#pragma unroll
    for (int g = 0; g < 8; g++) {
      tm[g] = fmaxf(fmaxf(s[0][g], s[1][g]), fmaxf(s[2][g], s[3][g]));
#pragma unroll
      for (int off = 8; off >= 1; off >>= 1)
        tm[g] = fmaxf(tm[g], __shfl_xor(tm[g], off, 32));
    }
    float rs[8];
#pragma unroll
    for (int g = 0; g < 8; g++) {
      float mnew  = fmaxf(m_run[g], tm[g]);
      float alpha = __expf(m_run[g] - mnew);
      l_run[g] *= alpha;
      m_run[g]  = mnew;
#pragma unroll
      for (int jn = 0; jn < 4; jn++) oacc[jn][g] *= alpha;
      rs[g] = 0.f;
    }
#pragma unroll
    for (int j = 0; j < 4; j++)
#pragma unroll
      for (int g = 0; g < 8; g++) {
        float p = __expf(s[j][g] - m_run[g]);
        rs[g] += p;
        int r = (lane < 16) ? g : g + 8;
        Pw[r * 72 + j * 16 + (lane & 15)] = (_Float16)p;
      }
#pragma unroll
    for (int g = 0; g < 8; g++) {
#pragma unroll
      for (int off = 8; off >= 1; off >>= 1) rs[g] += __shfl_xor(rs[g], off, 32);
      l_run[g] += rs[g];
    }

    // P is private per wave; DS ops are in-order per wave, wait to be safe.
    asm volatile("s_wait_dscnt 0x0" ::: "memory");

    // O += P V   (A frags from Pw, B frags from Vt)
#pragma unroll
    for (int tt = 0; tt < 2; tt++) {
      FragH ap;
      int r  = lane & 15;
      int o1 = tt * 32 + ((lane >> 4) << 3);
      ap.q[0] = *(const uint4*)&Pw[r * 72 + o1];
      ap.q[1] = *(const uint4*)&Pw[r * 72 + o1 + 16];
#pragma unroll
      for (int jn = 0; jn < 4; jn++) {
        FragH bv;
        int d  = jn * 16 + (lane & 15);
        int ko = tt * 32 + ((lane >> 4) << 3);
        bv.q[0] = *(const uint4*)&Vtlds[d * 72 + ko];
        bv.q[1] = *(const uint4*)&Vtlds[d * 72 + ko + 16];
        oacc[jn] = wmma_f16(ap.v, bv.v, oacc[jn]);
      }
    }
  }

  // epilogue: O /= l, write f16 to X[b, q, h*64+d]
  int qrow = qt * 128 + w * 16;
#pragma unroll
  for (int jn = 0; jn < 4; jn++) {
    int dcol = h * HD_ + jn * 16 + (lane & 15);
#pragma unroll
    for (int g = 0; g < 8; g++) {
      int r = (lane < 16) ? g : g + 8;
      float val = oacc[jn][g] / l_run[g];
      X[(size_t)(b * S_ + qrow + r) * D_ + dcol] = (_Float16)val;
    }
  }
}

// ---------------------------------------------------------------------------
extern "C" void kernel_launch(void* const* d_in, const int* in_sizes, int n_in,
                              void* d_out, int out_size, void* d_ws, size_t ws_size,
                              hipStream_t stream) {
  (void)in_sizes; (void)n_in; (void)out_size; (void)ws_size;
  const float* inputs_q  = (const float*)d_in[0];
  const float* inputs_kv = (const float*)d_in[1];
  const float* pos_q = (const float*)d_in[2];
  const float* pos_k = (const float*)d_in[3];
  const float* pos_v = (const float*)d_in[4];
  const float* mask  = (const float*)d_in[5];
  const float* wq = (const float*)d_in[6];
  const float* bq = (const float*)d_in[7];
  const float* wk = (const float*)d_in[8];
  const float* bk = (const float*)d_in[9];
  const float* wv = (const float*)d_in[10];
  const float* bv = (const float*)d_in[11];
  const float* wo = (const float*)d_in[12];
  const float* bo = (const float*)d_in[13];
  float* out = (float*)d_out;

  char* ws = (char*)d_ws;
  const size_t MB = 1024 * 1024;
  _Float16* qin = (_Float16*)(ws + 0 * MB);    // [4096,1024] f16
  _Float16* kin = (_Float16*)(ws + 8 * MB);
  _Float16* vin = (_Float16*)(ws + 16 * MB);
  _Float16* wtq = (_Float16*)(ws + 24 * MB);   // [N,K] f16
  _Float16* wtk = (_Float16*)(ws + 26 * MB);
  _Float16* wtv = (_Float16*)(ws + 28 * MB);
  _Float16* wto = (_Float16*)(ws + 30 * MB);
  _Float16* Qp  = (_Float16*)(ws + 32 * MB);   // [B,H,S,HD] f16
  _Float16* Kp  = (_Float16*)(ws + 40 * MB);
  _Float16* Vp  = (_Float16*)(ws + 48 * MB);
  _Float16* Xb  = (_Float16*)(ws + 56 * MB);   // [B*S, D] f16

  prep_qkv<<<4096, 256, 0, stream>>>(inputs_q, inputs_kv, pos_q, pos_k, pos_v,
                                     qin, kin, vin);
  dim3 tb(32, 8), tg(32, 32);
  transpose_w<<<tg, tb, 0, stream>>>(wq, wtq);
  transpose_w<<<tg, tb, 0, stream>>>(wk, wtk);
  transpose_w<<<tg, tb, 0, stream>>>(wv, wtv);
  transpose_w<<<tg, tb, 0, stream>>>(wo, wto);

  dim3 gg(8, 32);  // N/128, M/128
  gemm_f16<0><<<gg, 256, 0, stream>>>(qin, wtq, bq, 0.125f, Qp, nullptr);
  gemm_f16<0><<<gg, 256, 0, stream>>>(kin, wtk, bk, 1.0f,   Kp, nullptr);
  gemm_f16<0><<<gg, 256, 0, stream>>>(vin, wtv, bv, 1.0f,   Vp, nullptr);

  attn_fused<<<dim3(8, H_, B_), 256, 0, stream>>>(Qp, Kp, Vp, mask, Xb);

  gemm_f16<1><<<gg, 256, 0, stream>>>(Xb, wto, bo, 1.0f, nullptr, out);
}
